// VAE_Decoder_970662608903
// MI455X (gfx1250) — matrix operand order repaired and checked by reading, not compiled, not run on hardware
//
#include <hip/hip_runtime.h>
#include <hip/hip_bf16.h>

#define S_ 256
#define B_ 64
#define H_ 1024
#define R_ 256
#define O_ 256
#define G4H 4096     // 4*H
#define D0_ 1280     // R + H
#define D1_ 2048     // 2*H

typedef __attribute__((ext_vector_type(16))) __bf16 v16bf;
typedef __attribute__((ext_vector_type(8)))  float  v8f;

__device__ __forceinline__ v8f wmma_bf16(v16bf a, v16bf b, v8f c) {
  // D = A(16x32 bf16) x B(32x16 bf16) + C(16x16 f32)
  return __builtin_amdgcn_wmma_f32_16x16x32_bf16(false, a, false, b, (short)0, c, false, false);
}
__device__ __forceinline__ float sigmoidf_(float x) { return 1.0f / (1.0f + __expf(-x)); }

// ---------------------------------------------------------------------------
// Weight prep: pack a row-major [N][K] f32 matrix as bf16 WMMA-B tiles.
// Tile (nt,kt) covers B[k][n] = W[n][k] for n in [nt*16,+16), k in [kt*32,+32).
// In-tile layout matches the v16bf register image: lane l (0..31) holds
// column n = nt*16 + (l&15), k = kt*32 + (i>>3)*16 + (l>>4)*8 + (i&7) for i = 0..15,
// stored lane-major (512 bf16 per tile) so each lane reads 32 contiguous bytes.
// ---------------------------------------------------------------------------
__global__ void pack_b_kernel(const float* __restrict__ W, __bf16* __restrict__ P,
                              int N, int K) {
  long long total = (long long)N * K;
  long long e = (long long)blockIdx.x * blockDim.x + threadIdx.x;
  if (e >= total) return;
  int KT = K >> 5;
  int i = (int)(e & 15);
  int l = (int)((e >> 4) & 31);
  long long tile = e >> 9;
  int kt = (int)(tile % KT);
  long long nt = tile / KT;
  long long n = nt * 16 + (l & 15);
  int k = kt * 32 + ((i >> 3) << 4) + ((l >> 4) << 3) + (i & 7);
  P[e] = (__bf16)W[n * K + k];
}

// WxT[d][r][j] = W_ih0[d][j][r]  (one-hot slab, transposed for coalesced gather)
__global__ void wxt_kernel(const float* __restrict__ Wih0, float* __restrict__ WxT) {
  int gid = blockIdx.x * blockDim.x + threadIdx.x;       // 2*256*4096
  if (gid >= 2 * R_ * G4H) return;
  int j = gid & (G4H - 1);
  int r = (gid >> 12) & (R_ - 1);
  int d = gid >> 20;
  WxT[gid] = Wih0[((long long)(d * G4H + j)) * D0_ + r];
}

// condvec[d][b][j] = b0[d][j] + sum_k emb[cond[b]][k] * W_ih0[d][j][R+k]
__global__ void condvec_kernel(const float* __restrict__ Wih0, const float* __restrict__ b0,
                               const float* __restrict__ emb, const int* __restrict__ cond,
                               float* __restrict__ condv) {
  int gid = blockIdx.x * blockDim.x + threadIdx.x;       // 2*64*4096
  if (gid >= 2 * B_ * G4H) return;
  int j = gid & (G4H - 1);
  int b = (gid >> 12) & (B_ - 1);
  int d = gid >> 18;
  int ci = cond[b];
  const float* wrow = Wih0 + ((long long)(d * G4H + j)) * D0_ + R_;
  const float* ev = emb + (long long)ci * H_;
  float acc = b0[d * G4H + j];
  for (int k = 0; k < H_; k += 4) {
    acc += ev[k] * wrow[k] + ev[k + 1] * wrow[k + 1] +
           ev[k + 2] * wrow[k + 2] + ev[k + 3] * wrow[k + 3];
  }
  condv[gid] = acc;
}

__global__ void zero_kernel(float* __restrict__ p, int n) {
  int g = blockIdx.x * blockDim.x + threadIdx.x;
  if (g < n) p[g] = 0.0f;
}

// ---------------------------------------------------------------------------
// Shared GEMM inner loop: acc[4 mtiles] += A(64 x 64*nchunks) x Bpacked.
// Block = 256 threads / 8 waves; A (f32) is streamed through LDS in 64-wide K
// chunks, converted to bf16 in the exact WMMA-A register image.
// ---------------------------------------------------------------------------
__device__ __forceinline__ void gemm_bf16_acc(
    v8f acc[4], const float* __restrict__ A, int lda, int nchunks,
    const __bf16* __restrict__ BP, int KT, int nt_full, __bf16* smA)
{
  const int tid = threadIdx.x;
  const int lane = tid & 31;
  const int lan15 = lane & 15;
  const int lhi8 = (lane >> 4) << 3;       // 0 or 8 (A-layout K sub-offset)
  const int stile = tid >> 5;              // 0..7 = mtile*2 + ktc
  const int smt = stile >> 1;
  const int sktc = stile & 1;
  __bf16* sdst = smA + stile * 512 + lane * 16;
  const float* srcbase = A + (long long)(smt * 16 + lan15) * lda + lhi8 + sktc * 32;
  for (int chunk = 0; chunk < nchunks; ++chunk) {
    __syncthreads();                       // WAR on smA vs previous iteration
    const float* src = srcbase + chunk * 64;
#pragma unroll
    for (int i = 0; i < 8; ++i) sdst[i] = (__bf16)src[i];
#pragma unroll
    for (int i = 0; i < 8; ++i) sdst[8 + i] = (__bf16)src[16 + i];
    __syncthreads();
#pragma unroll
    for (int ktc = 0; ktc < 2; ++ktc) {
      int kt = chunk * 2 + ktc;
      v16bf bf = *reinterpret_cast<const v16bf*>(
          BP + (((long long)nt_full * KT + kt) << 9) + lane * 16);
#pragma unroll
      for (int mt = 0; mt < 4; ++mt) {
        v16bf af = *reinterpret_cast<const v16bf*>(smA + (mt * 2 + ktc) * 512 + lane * 16);
        acc[mt] = wmma_bf16(af, bf, acc[mt]);
      }
    }
  }
}

// ---------------------------------------------------------------------------
// Layer-0 LSTM step: grid 64 = 2 dirs x 32 hidden-slices (32 units each).
// Gate preactivation init = condvec + one-hot column gather (no input GEMM!),
// then recurrent GEMM (K=1024) via WMMA, then gate nonlinearity.
// ---------------------------------------------------------------------------
__global__ __launch_bounds__(256) void lstm0_step_kernel(
    const float* __restrict__ h_prev, float* __restrict__ h_next,
    float* __restrict__ c, float* __restrict__ y0_t,
    const float* __restrict__ condv, const float* __restrict__ WxT,
    const __bf16* __restrict__ WhhP, const int* __restrict__ e_t)
{
  __shared__ __align__(32) __bf16 smA[8 * 512];
  __shared__ float smG[4][B_][32];

  const int tid = threadIdx.x;
  const int lane = tid & 31;
  const int w = tid >> 5;
  const int d = blockIdx.x >> 5;
  const int blk = blockIdx.x & 31;
  const int m0 = blk << 5;
  const int gi = w >> 1;                   // gate 0..3 (i,f,g,o)
  const int half = w & 1;
  const int n_base = gi * H_ + m0 + (half << 4);
  const int nt_full = (d << 8) + (gi << 6) + (blk << 1) + half;
  const int lan15 = lane & 15;
  const int lhi8 = (lane >> 4) << 3;

  v8f acc[4];
  {
    int n = n_base + lan15;
    const float* cv = condv + (long long)d * B_ * G4H + n;
#pragma unroll
    for (int mt = 0; mt < 4; ++mt) {
      v8f a;
#pragma unroll
      for (int r = 0; r < 8; ++r) {
        int brow = mt * 16 + r + lhi8;
        int e = e_t[brow];
        a[r] = cv[(long long)brow * G4H] + WxT[((long long)(d * R_ + e)) * G4H + n];
      }
      acc[mt] = a;
    }
  }

  gemm_bf16_acc(acc, h_prev + (long long)d * B_ * H_, H_, H_ / 64, WhhP, H_ / 32, nt_full, smA);

#pragma unroll
  for (int mt = 0; mt < 4; ++mt)
#pragma unroll
    for (int r = 0; r < 8; ++r)
      smG[gi][mt * 16 + r + lhi8][(half << 4) + lan15] = acc[mt][r];
  __syncthreads();

  float* cd = c + (long long)d * B_ * H_;
  float* hn = h_next + (long long)d * B_ * H_;
#pragma unroll
  for (int it = 0; it < 8; ++it) {
    int idx = it * 256 + tid;
    int b = idx >> 5;
    int ml = idx & 31;
    float gI = smG[0][b][ml], gF = smG[1][b][ml], gG = smG[2][b][ml], gO = smG[3][b][ml];
    long long ci = (long long)b * H_ + m0 + ml;
    float cn = sigmoidf_(gF) * cd[ci] + sigmoidf_(gI) * tanhf(gG);
    float hh = sigmoidf_(gO) * tanhf(cn);
    cd[ci] = cn;
    hn[ci] = hh;
    y0_t[(long long)b * D1_ + d * H_ + m0 + ml] = hh;
  }
}

// ---------------------------------------------------------------------------
// Layer-1 LSTM step: bias init, GEMM over y0[t] (K=2048) + recurrent (K=1024).
// ---------------------------------------------------------------------------
__global__ __launch_bounds__(256) void lstm1_step_kernel(
    const float* __restrict__ h_prev, float* __restrict__ h_next,
    float* __restrict__ c, float* __restrict__ y1_t,
    const float* __restrict__ y0_t, const float* __restrict__ b1,
    const __bf16* __restrict__ WihP, const __bf16* __restrict__ WhhP)
{
  __shared__ __align__(32) __bf16 smA[8 * 512];
  __shared__ float smG[4][B_][32];

  const int tid = threadIdx.x;
  const int lane = tid & 31;
  const int w = tid >> 5;
  const int d = blockIdx.x >> 5;
  const int blk = blockIdx.x & 31;
  const int m0 = blk << 5;
  const int gi = w >> 1;
  const int half = w & 1;
  const int n_base = gi * H_ + m0 + (half << 4);
  const int nt_full = (d << 8) + (gi << 6) + (blk << 1) + half;
  const int lan15 = lane & 15;
  const int lhi8 = (lane >> 4) << 3;

  v8f acc[4];
  {
    float bb = b1[d * G4H + n_base + lan15];
#pragma unroll
    for (int mt = 0; mt < 4; ++mt) {
      v8f a;
#pragma unroll
      for (int r = 0; r < 8; ++r) a[r] = bb;
      acc[mt] = a;
    }
  }

  gemm_bf16_acc(acc, y0_t, D1_, D1_ / 64, WihP, D1_ / 32, nt_full, smA);
  gemm_bf16_acc(acc, h_prev + (long long)d * B_ * H_, H_, H_ / 64, WhhP, H_ / 32, nt_full, smA);

#pragma unroll
  for (int mt = 0; mt < 4; ++mt)
#pragma unroll
    for (int r = 0; r < 8; ++r)
      smG[gi][mt * 16 + r + lhi8][(half << 4) + lan15] = acc[mt][r];
  __syncthreads();

  float* cd = c + (long long)d * B_ * H_;
  float* hn = h_next + (long long)d * B_ * H_;
#pragma unroll
  for (int it = 0; it < 8; ++it) {
    int idx = it * 256 + tid;
    int b = idx >> 5;
    int ml = idx & 31;
    float gI = smG[0][b][ml], gF = smG[1][b][ml], gG = smG[2][b][ml], gO = smG[3][b][ml];
    long long ci = (long long)b * H_ + m0 + ml;
    float cn = sigmoidf_(gF) * cd[ci] + sigmoidf_(gI) * tanhf(gG);
    float hh = sigmoidf_(gO) * tanhf(cn);
    cd[ci] = cn;
    hn[ci] = hh;
    y1_t[(long long)b * D1_ + d * H_ + m0 + ml] = hh;
  }
}

// ---------------------------------------------------------------------------
// FC: out[16384,256] = y1[16384,2048] x fc_W^T + fc_b.  256 blocks x 64 rows.
// Each wave: 2 N-tiles x 4 M-tiles.
// ---------------------------------------------------------------------------
__global__ __launch_bounds__(256) void fc_kernel(
    const float* __restrict__ y1, const __bf16* __restrict__ fcP,
    const float* __restrict__ fc_b, float* __restrict__ out)
{
  __shared__ __align__(32) __bf16 smA[8 * 512];
  const int tid = threadIdx.x;
  const int lane = tid & 31;
  const int w = tid >> 5;
  const int lan15 = lane & 15;
  const int lhi8 = (lane >> 4) << 3;
  const long long rowbase = (long long)blockIdx.x * 64;

  v8f acc[4][2];
#pragma unroll
  for (int j = 0; j < 2; ++j) {
    float bb = fc_b[(2 * w + j) * 16 + lan15];
#pragma unroll
    for (int mt = 0; mt < 4; ++mt) {
      v8f a;
#pragma unroll
      for (int r = 0; r < 8; ++r) a[r] = bb;
      acc[mt][j] = a;
    }
  }

  const int stile = tid >> 5;
  const int smt = stile >> 1;
  const int sktc = stile & 1;
  __bf16* sdst = smA + stile * 512 + lane * 16;
  const float* srcbase = y1 + (rowbase + smt * 16 + lan15) * D1_ + lhi8 + sktc * 32;

  for (int chunk = 0; chunk < D1_ / 64; ++chunk) {
    __syncthreads();
    const float* src = srcbase + chunk * 64;
#pragma unroll
    for (int i = 0; i < 8; ++i) sdst[i] = (__bf16)src[i];
#pragma unroll
    for (int i = 0; i < 8; ++i) sdst[8 + i] = (__bf16)src[16 + i];
    __syncthreads();
#pragma unroll
    for (int ktc = 0; ktc < 2; ++ktc) {
      int kt = chunk * 2 + ktc;
#pragma unroll
      for (int j = 0; j < 2; ++j) {
        int nt = 2 * w + j;
        v16bf bf = *reinterpret_cast<const v16bf*>(
            fcP + (((long long)nt * (D1_ / 32) + kt) << 9) + lane * 16);
#pragma unroll
        for (int mt = 0; mt < 4; ++mt) {
          v16bf af = *reinterpret_cast<const v16bf*>(smA + (mt * 2 + ktc) * 512 + lane * 16);
          acc[mt][j] = wmma_bf16(af, bf, acc[mt][j]);
        }
      }
    }
  }

#pragma unroll
  for (int mt = 0; mt < 4; ++mt)
#pragma unroll
    for (int j = 0; j < 2; ++j)
#pragma unroll
      for (int r = 0; r < 8; ++r)
        out[(rowbase + mt * 16 + r + lhi8) * O_ + (2 * w + j) * 16 + lan15] = acc[mt][j][r];
}

// Per-row NLL of log_softmax; one wave per row (deterministic lane reduction).
__global__ __launch_bounds__(256) void loss_rows_kernel(
    const float* __restrict__ out, const int* __restrict__ e_flat,
    float* __restrict__ rowloss)
{
  const int w = threadIdx.x >> 5;
  const int lane = threadIdx.x & 31;
  const long long row = (long long)blockIdx.x * 8 + w;
  const float* x = out + row * O_;
  float v[8];
  float m = -3.0e38f;
#pragma unroll
  for (int i = 0; i < 8; ++i) { v[i] = x[lane + i * 32]; m = fmaxf(m, v[i]); }
#pragma unroll
  for (int off = 16; off; off >>= 1) m = fmaxf(m, __shfl_xor(m, off));
  float s = 0.0f;
#pragma unroll
  for (int i = 0; i < 8; ++i) s += __expf(v[i] - m);
#pragma unroll
  for (int off = 16; off; off >>= 1) s += __shfl_xor(s, off);
  if (lane == 0) {
    int lbl = e_flat[row];
    rowloss[row] = -(x[lbl] - m - logf(s));
  }
}

// Fixed-order final reduction -> deterministic scalar loss.
__global__ __launch_bounds__(256) void loss_reduce_kernel(
    const float* __restrict__ rowloss, float* __restrict__ lossout)
{
  __shared__ float sm[256];
  const int tid = threadIdx.x;
  float s = 0.0f;
  for (int i = 0; i < 64; ++i) s += rowloss[tid * 64 + i];
  sm[tid] = s;
  __syncthreads();
  for (int k = 128; k > 0; k >>= 1) {
    if (tid < k) sm[tid] += sm[tid + k];
    __syncthreads();
  }
  if (tid == 0) *lossout = sm[0] * (1.0f / 16384.0f);
}

__global__ void final_state_kernel(const float* __restrict__ h0f, const float* __restrict__ c0,
                                   const float* __restrict__ h1f, const float* __restrict__ c1,
                                   float* __restrict__ outh, float* __restrict__ outc)
{
  int idx = blockIdx.x * blockDim.x + threadIdx.x;       // 4*64*1024
  if (idx >= 4 * B_ * H_) return;
  int ld = idx >> 16;
  int within = idx & 65535;
  const float* hs = (ld < 2) ? h0f : h1f;
  const float* cs = (ld < 2) ? c0 : c1;
  outh[idx] = hs[(ld & 1) * (B_ * H_) + within];
  outc[idx] = cs[(ld & 1) * (B_ * H_) + within];
}

// ---------------------------------------------------------------------------
extern "C" void kernel_launch(void* const* d_in, const int* in_sizes, int n_in,
                              void* d_out, int out_size, void* d_ws, size_t ws_size,
                              hipStream_t stream)
{
  (void)in_sizes; (void)n_in; (void)out_size; (void)ws_size;
  const int*   cond    = (const int*)d_in[1];
  const int*   e_input = (const int*)d_in[2];
  const float* emb     = (const float*)d_in[3];
  const float* W_ih0   = (const float*)d_in[4];
  const float* W_hh0   = (const float*)d_in[5];
  const float* b0      = (const float*)d_in[6];
  const float* W_ih1   = (const float*)d_in[7];
  const float* W_hh1   = (const float*)d_in[8];
  const float* b1      = (const float*)d_in[9];
  const float* fc_W    = (const float*)d_in[10];
  const float* fc_b    = (const float*)d_in[11];
  float* out = (float*)d_out;

  char* ws = (char*)d_ws;
  size_t off = 0;
  auto alloc = [&](size_t bytes) -> char* {
    char* p = ws + off;
    off += (bytes + 255) & ~(size_t)255;
    return p;
  };
  __bf16* WhhP0 = (__bf16*)alloc((size_t)2 * G4H * H_ * 2);
  __bf16* WihP1 = (__bf16*)alloc((size_t)2 * G4H * D1_ * 2);
  __bf16* WhhP1 = (__bf16*)alloc((size_t)2 * G4H * H_ * 2);
  __bf16* fcP   = (__bf16*)alloc((size_t)O_ * D1_ * 2);
  float* WxT    = (float*)alloc((size_t)2 * R_ * G4H * 4);
  float* condv  = (float*)alloc((size_t)2 * B_ * G4H * 4);
  float* h0A    = (float*)alloc((size_t)2 * B_ * H_ * 4);
  float* h0B    = (float*)alloc((size_t)2 * B_ * H_ * 4);
  float* c0     = (float*)alloc((size_t)2 * B_ * H_ * 4);
  float* h1A    = (float*)alloc((size_t)2 * B_ * H_ * 4);
  float* h1B    = (float*)alloc((size_t)2 * B_ * H_ * 4);
  float* c1     = (float*)alloc((size_t)2 * B_ * H_ * 4);
  float* y0     = (float*)alloc((size_t)S_ * B_ * D1_ * 4);
  float* y1     = (float*)alloc((size_t)S_ * B_ * D1_ * 4);
  float* rowloss= (float*)alloc((size_t)S_ * B_ * 4);

  // --- weight prep ---
  pack_b_kernel<<<(2 * G4H * H_ + 255) / 256, 256, 0, stream>>>(W_hh0, WhhP0, 2 * G4H, H_);
  pack_b_kernel<<<(2 * G4H * D1_ + 255) / 256, 256, 0, stream>>>(W_ih1, WihP1, 2 * G4H, D1_);
  pack_b_kernel<<<(2 * G4H * H_ + 255) / 256, 256, 0, stream>>>(W_hh1, WhhP1, 2 * G4H, H_);
  pack_b_kernel<<<(O_ * D1_ + 255) / 256, 256, 0, stream>>>(fc_W, fcP, O_, D1_);
  wxt_kernel<<<(2 * R_ * G4H) / 256, 256, 0, stream>>>(W_ih0, WxT);
  condvec_kernel<<<(2 * B_ * G4H) / 256, 256, 0, stream>>>(W_ih0, b0, emb, cond, condv);

  const int st = 2 * B_ * H_;
  zero_kernel<<<(st + 255) / 256, 256, 0, stream>>>(h0A, st);
  zero_kernel<<<(st + 255) / 256, 256, 0, stream>>>(h1A, st);
  zero_kernel<<<(st + 255) / 256, 256, 0, stream>>>(c0, st);
  zero_kernel<<<(st + 255) / 256, 256, 0, stream>>>(c1, st);

  // --- layer 0 recurrence (both "directions" scan forward, per reference quirk) ---
  for (int t = 0; t < S_; ++t) {
    const float* hp = (t & 1) ? h0B : h0A;
    float* hn = (t & 1) ? h0A : h0B;
    lstm0_step_kernel<<<64, 256, 0, stream>>>(
        hp, hn, c0, y0 + (size_t)t * B_ * D1_, condv, WxT, WhhP0, e_input + t * B_);
  }
  // --- layer 1 recurrence ---
  for (int t = 0; t < S_; ++t) {
    const float* hp = (t & 1) ? h1B : h1A;
    float* hn = (t & 1) ? h1A : h1B;
    lstm1_step_kernel<<<64, 256, 0, stream>>>(
        hp, hn, c1, y1 + (size_t)t * B_ * D1_, y0 + (size_t)t * B_ * D1_, b1, WihP1, WhhP1);
  }

  // --- FC + loss + final states ---
  fc_kernel<<<(S_ * B_) / 64, 256, 0, stream>>>(y1, fcP, fc_b, out);
  loss_rows_kernel<<<(S_ * B_) / 8, 256, 0, stream>>>(out, e_input, rowloss);

  float* hN = out + (size_t)S_ * B_ * O_;                 // 4194304
  float* cN = hN + (size_t)4 * B_ * H_;                   // +262144
  float* lossp = cN + (size_t)4 * B_ * H_;                // +262144
  loss_reduce_kernel<<<1, 256, 0, stream>>>(rowloss, lossp);
  // after t=255 (odd), the final h lives in the "A" buffers
  final_state_kernel<<<(4 * B_ * H_) / 256, 256, 0, stream>>>(h0A, c0, h1A, c1, hN, cN);
}
